// LRUModel_77068893160294
// MI455X (gfx1250) — compile-verified, matrix operand order checked by reading, not run on hardware
//
#include <hip/hip_runtime.h>

typedef __attribute__((ext_vector_type(16))) _Float16 v16h;
typedef __attribute__((ext_vector_type(8)))  _Float16 v8h;
typedef __attribute__((ext_vector_type(4)))  _Float16 v4h;
typedef __attribute__((ext_vector_type(8)))  float    v8f;

#define HID    64
#define XDIM   128
#define SEQ    48
#define MSLOT  4

// LDS row pitches in _Float16 units (chosen so row*pitch*2 is a multiple of 16B
// and rows stagger across banks: 272B and 144B pitches).
#define XP   136   // X tile: 128 rows x (128 + 8 pad)
#define W1P  136   // W1: 64 rows x (128 + 8 pad)
#define W2P  72    // W2: 64 rows x (64 + 8 pad)
#define HP   72    // per-wave H tile: 16 rows x (64 + 8 pad)

__global__ __launch_bounds__(256, 2) void lru_fused_wmma(
    const int*   __restrict__ seqs,
    const int*   __restrict__ qtok,
    const float* __restrict__ embed,
    const float* __restrict__ W1,
    const float* __restrict__ b1,
    const float* __restrict__ W2,
    const float* __restrict__ b2,
    float*       __restrict__ out)
{
    __shared__ __align__(16) _Float16 Xs [128 * XP];
    __shared__ __align__(16) _Float16 W1s[ 64 * W1P];
    __shared__ __align__(16) _Float16 W2s[ 64 * W2P];
    __shared__ __align__(16) _Float16 Hs [  8 * 16 * HP];

    const int t         = threadIdx.x;
    const int block_row0 = blockIdx.x * 128;

    // ---------------- Phase 1a: stage X tile (gather + f32->f16) ----------
    // threads 0..127: query-embedding half (cols 0..63) of row (t)
    // threads 128..255: memory-mean half (cols 64..127) of row (t-128)
    {
        const int r   = t & 127;
        const int row = block_row0 + r;
        if (t < 128) {
            const int q = qtok[row];
            const float4* e = (const float4*)(embed + q * HID);
            _Float16* dst = Xs + r * XP;
            #pragma unroll
            for (int i = 0; i < 16; ++i) {
                float4 v = e[i];
                v4h h;
                h[0] = (_Float16)v.x; h[1] = (_Float16)v.y;
                h[2] = (_Float16)v.z; h[3] = (_Float16)v.w;
                *(v4h*)(dst + 4 * i) = h;
            }
        } else {
            const int* s = seqs + row * SEQ + (SEQ - 1 - MSLOT);
            const int t0 = s[0], t1 = s[1], t2 = s[2], t3 = s[3];
            const float4* e0 = (const float4*)(embed + t0 * HID);
            const float4* e1 = (const float4*)(embed + t1 * HID);
            const float4* e2 = (const float4*)(embed + t2 * HID);
            const float4* e3 = (const float4*)(embed + t3 * HID);
            _Float16* dst = Xs + r * XP + HID;
            #pragma unroll
            for (int i = 0; i < 16; ++i) {
                float4 a = e0[i], b = e1[i], c = e2[i], d = e3[i];
                v4h h;
                h[0] = (_Float16)(0.25f * (a.x + b.x + c.x + d.x));
                h[1] = (_Float16)(0.25f * (a.y + b.y + c.y + d.y));
                h[2] = (_Float16)(0.25f * (a.z + b.z + c.z + d.z));
                h[3] = (_Float16)(0.25f * (a.w + b.w + c.w + d.w));
                *(v4h*)(dst + 4 * i) = h;
            }
        }
    }

    // ---------------- Phase 1b: stage weights as f16 [j][k] ----------------
    // W1: 64x128 f32 = 2048 float4, 8 iters per thread
    for (int idx = t; idx < 2048; idx += 256) {
        float4 v = ((const float4*)W1)[idx];
        const int flat = idx * 4;
        const int j = flat >> 7;
        const int k = flat & 127;
        v4h h;
        h[0] = (_Float16)v.x; h[1] = (_Float16)v.y;
        h[2] = (_Float16)v.z; h[3] = (_Float16)v.w;
        *(v4h*)(W1s + j * W1P + k) = h;
    }
    // W2: 64x64 f32 = 1024 float4, 4 iters per thread
    for (int idx = t; idx < 1024; idx += 256) {
        float4 v = ((const float4*)W2)[idx];
        const int flat = idx * 4;
        const int j = flat >> 6;
        const int k = flat & 63;
        v4h h;
        h[0] = (_Float16)v.x; h[1] = (_Float16)v.y;
        h[2] = (_Float16)v.z; h[3] = (_Float16)v.w;
        *(v4h*)(W2s + j * W2P + k) = h;
    }

    __syncthreads();

    // ---------------- Phase 2: per-wave WMMA compute -----------------------
    const int lane  = t & 31;        // wave32 lane
    const int wave  = t >> 5;        // 0..7, 16 rows each
    const int lhalf = lane >> 4;     // K-half select per ISA 16-bit A/B layout
    const int l15   = lane & 15;

    // Build a v16h A/B fragment from LDS: lane<16 -> K = kb+0..7, kb+16..23;
    // lane>=16 -> K = kb+8..15, kb+24..31 (ISA 7.12.2, 16-bit 16x32 layout).
    auto loadFrag = [&](const _Float16* base, int pitch, int row, int kb) -> v16h {
        const _Float16* p = base + row * pitch + kb + lhalf * 8;
        v8h lo = *(const v8h*)(p);        // 16B ds_load_b128
        v8h hi = *(const v8h*)(p + 16);   // 16B ds_load_b128
        v16h f;
        #pragma unroll
        for (int i = 0; i < 8; ++i) { f[i] = lo[i]; f[8 + i] = hi[i]; }
        return f;
    };

    // ----- Layer 1: H(16x64) = X(16x128) @ W1^T, as 4 Ktiles x 4 Ntiles ----
    v8f acc[4];
    #pragma unroll
    for (int nt = 0; nt < 4; ++nt)
        #pragma unroll
        for (int i = 0; i < 8; ++i) acc[nt][i] = 0.0f;

    #pragma unroll
    for (int kf = 0; kf < 4; ++kf) {
        const int kb = kf * 32;
        v16h a = loadFrag(Xs, XP, wave * 16 + l15, kb);
        #pragma unroll
        for (int nt = 0; nt < 4; ++nt) {
            v16h b = loadFrag(W1s, W1P, nt * 16 + l15, kb);
            acc[nt] = __builtin_amdgcn_wmma_f32_16x16x32_f16(
                false, a, false, b, (short)0, acc[nt], false, false);
        }
    }

    // bias + relu, spill H to per-wave LDS tile (C-layout -> A-layout reshape)
    _Float16* hs = Hs + wave * 16 * HP;
    #pragma unroll
    for (int nt = 0; nt < 4; ++nt) {
        const float bv = b1[nt * 16 + l15];
        #pragma unroll
        for (int r = 0; r < 8; ++r) {
            float h = acc[nt][r] + bv;
            h = h > 0.0f ? h : 0.0f;
            const int m = lhalf * 8 + r;          // C-layout row
            hs[m * HP + nt * 16 + l15] = (_Float16)h;
        }
    }

    // ----- Layer 2: OUT(16x64) = H(16x64) @ W2^T, 2 Ktiles x 4 Ntiles ------
    v8f acc2[4];
    #pragma unroll
    for (int nt = 0; nt < 4; ++nt)
        #pragma unroll
        for (int i = 0; i < 8; ++i) acc2[nt][i] = 0.0f;

    #pragma unroll
    for (int kf = 0; kf < 2; ++kf) {
        const int kb = kf * 32;
        v16h a = loadFrag(hs, HP, l15, kb);
        #pragma unroll
        for (int nt = 0; nt < 4; ++nt) {
            v16h b = loadFrag(W2s, W2P, nt * 16 + l15, kb);
            acc2[nt] = __builtin_amdgcn_wmma_f32_16x16x32_f16(
                false, a, false, b, (short)0, acc2[nt], false, false);
        }
    }

    // ----- bias + store f32 output ----------------------------------------
    #pragma unroll
    for (int nt = 0; nt < 4; ++nt) {
        const float bv = b2[nt * 16 + l15];
        #pragma unroll
        for (int r = 0; r < 8; ++r) {
            const int m   = lhalf * 8 + r;
            const int row = block_row0 + wave * 16 + m;
            out[row * HID + nt * 16 + l15] = acc2[nt][r] + bv;
        }
    }
}

extern "C" void kernel_launch(void* const* d_in, const int* in_sizes, int n_in,
                              void* d_out, int out_size, void* d_ws, size_t ws_size,
                              hipStream_t stream) {
    const int*   seqs  = (const int*)  d_in[0];
    const int*   qtok  = (const int*)  d_in[1];
    const float* embed = (const float*)d_in[2];
    const float* W1    = (const float*)d_in[3];
    const float* b1    = (const float*)d_in[4];
    const float* W2    = (const float*)d_in[5];
    const float* b2    = (const float*)d_in[6];
    float*       out   = (float*)d_out;

    const int B = in_sizes[1];          // query_tok count == batch (524288)
    const int blocks = B / 128;         // 128 rows per block -> 4096 blocks

    lru_fused_wmma<<<blocks, 256, 0, stream>>>(seqs, qtok, embed, W1, b1, W2, b2, out);
}
